// Aligner_66675072303793
// MI455X (gfx1250) — matrix operand order repaired and checked by reading, not compiled, not run on hardware
//
#include <hip/hip_runtime.h>
#include <hip/hip_bf16.h>
#include <stdint.h>

typedef __attribute__((ext_vector_type(16))) _Float16 v16h;
typedef __attribute__((ext_vector_type(2)))  _Float16 h2;
typedef __attribute__((ext_vector_type(8)))  float    v8f;
typedef __attribute__((ext_vector_type(4)))  float    f4;

#define B_      8
#define D_      512
#define T_      4096
#define L_      1024
#define STRIDE_ 4
#define SIGSQ   5.0f
#define NEGC    (-1000000000.0f)

// -------- d_out layout (reference return order, flattened as float32) --------
// z:        B*D*L = 4194304
// z_mask:   B*L   = 8192
// z_len:    B     = 8
// align:    B*L*T = 33554432
// loss:     1
#define OFF_Z      ((size_t)0)
#define OFF_ZMASK  ((size_t)4194304)
#define OFF_ZLEN   ((size_t)4202496)
#define OFF_ALIGN  ((size_t)4202504)
#define OFF_LOSS   ((size_t)37756936)

// ---------------------------------------------------------------------------
// 1) score[b,t] = exp(sum_d W[d]*x[b,d,t]) * mask
// ---------------------------------------------------------------------------
__global__ __launch_bounds__(256) void k_score(const float* __restrict__ x,
                                               const float* __restrict__ W,
                                               const uint8_t* __restrict__ mask,
                                               float* __restrict__ score) {
  int idx = blockIdx.x * 256 + threadIdx.x;           // over B*T
  int b = idx / T_, t = idx % T_;
  const float* xp = x + (size_t)b * D_ * T_ + t;
  float acc = 0.f;
#pragma unroll 8
  for (int d = 0; d < D_; ++d) acc += W[d] * xp[(size_t)d * T_];
  float m = mask[idx] ? 1.f : 0.f;
  score[idx] = __expf(acc) * m;
}

// ---------------------------------------------------------------------------
// 2) per-batch inclusive scan -> csn, plus score_loss partial per batch
//    csn[t] = (cum[t]-cum[0]) * (zl-1)/(cum[T-1]-cum[0])
//    loss_b = sum_{t>=1} relu(score[t]*scale - 1) / (len-1)
// ---------------------------------------------------------------------------
__global__ __launch_bounds__(1024) void k_scan(const float* __restrict__ score,
                                               const int* __restrict__ xlen,
                                               float* __restrict__ csn,
                                               float* __restrict__ loss_b) {
  __shared__ float arr[1024];
  int b = blockIdx.x;
  int i = threadIdx.x;
  const float* sp = score + (size_t)b * T_ + i * 4;
  float s0 = sp[0], s1 = sp[1], s2 = sp[2], s3 = sp[3];
  float c1 = s0 + s1, c2 = c1 + s2, c3 = c2 + s3;
  arr[i] = c3;
  __syncthreads();
  float v = c3;
  for (int off = 1; off < 1024; off <<= 1) {
    float u = (i >= off) ? arr[i - off] : 0.f;
    __syncthreads();
    v += u;
    arr[i] = v;
    __syncthreads();
  }
  float total = arr[1023];
  float excl  = v - c3;
  float c0    = score[(size_t)b * T_];
  float len   = (float)xlen[b];
  float zl    = ceilf(len / (float)STRIDE_);
  float scale = (zl - 1.f) / (total - c0);

  float cums[4] = {excl + s0, excl + c1, excl + c2, excl + c3};
  float* cp = csn + (size_t)b * T_ + i * 4;
#pragma unroll
  for (int j = 0; j < 4; ++j) cp[j] = (cums[j] - c0) * scale;

  float sv[4] = {s0, s1, s2, s3};
  float lp = 0.f;
#pragma unroll
  for (int j = 0; j < 4; ++j) {
    int t = i * 4 + j;
    if (t >= 1) lp += fmaxf(sv[j] * scale - 1.f, 0.f);
  }
  __syncthreads();
  arr[i] = lp;
  __syncthreads();
  for (int off = 512; off > 0; off >>= 1) {
    if (i < off) arr[i] += arr[i + off];
    __syncthreads();
  }
  if (i == 0) loss_b[b] = arr[0] / (len - 1.f);
}

// ---------------------------------------------------------------------------
// 3) alignment[b,l,t] = softmax_t( -5*(l-csn)^2, masked+causal ) * z_mask
//    one block per (b,l), 256 threads, 16 t's each (stride-256, coalesced)
// ---------------------------------------------------------------------------
__global__ __launch_bounds__(256) void k_align(const float* __restrict__ csn,
                                               const uint8_t* __restrict__ mask,
                                               float* __restrict__ align) {
  int bid = blockIdx.x;                 // over B*L
  int b = bid / L_, l = bid % L_;
  int tid = threadIdx.x;
  const float* cs = csn + (size_t)b * T_;
  const uint8_t* mk = mask + (size_t)b * T_;
  int tmax = (l + 1) * STRIDE_;

  float dist[16];
  float mx = -3.4e38f;
#pragma unroll
  for (int j = 0; j < 16; ++j) {
    int t = tid + j * 256;
    float dd = (float)l - cs[t];
    float d  = -SIGSQ * dd * dd;
    bool valid = (mk[t] != 0) && (t < tmax);
    d = valid ? d : NEGC;
    dist[j] = d;
    mx = fmaxf(mx, d);
  }
  __shared__ float redm[8];
  __shared__ float reds[8];
  for (int k = 16; k > 0; k >>= 1) mx = fmaxf(mx, __shfl_xor(mx, k, 32));
  if ((tid & 31) == 0) redm[tid >> 5] = mx;
  __syncthreads();
  mx = redm[0];
#pragma unroll
  for (int w = 1; w < 8; ++w) mx = fmaxf(mx, redm[w]);

  float ex[16];
  float sum = 0.f;
#pragma unroll
  for (int j = 0; j < 16; ++j) {
    ex[j] = __expf(dist[j] - mx);
    sum += ex[j];
  }
  for (int k = 16; k > 0; k >>= 1) sum += __shfl_xor(sum, k, 32);
  if ((tid & 31) == 0) reds[tid >> 5] = sum;
  __syncthreads();
  sum = 0.f;
#pragma unroll
  for (int w = 0; w < 8; ++w) sum += reds[w];

  float zm  = mk[l * STRIDE_] ? 1.f : 0.f;
  float inv = zm / sum;
  float* ap = align + ((size_t)b * L_ + l) * T_;
#pragma unroll
  for (int j = 0; j < 16; ++j) ap[tid + j * 256] = ex[j] * inv;
}

// ---------------------------------------------------------------------------
// 4) misc outputs: z_mask, z_lengths, score_loss mean
// ---------------------------------------------------------------------------
__global__ __launch_bounds__(256) void k_misc(const uint8_t* __restrict__ mask,
                                              const int* __restrict__ xlen,
                                              const float* __restrict__ loss_b,
                                              float* __restrict__ out_zmask,
                                              float* __restrict__ out_zlen,
                                              float* __restrict__ out_loss) {
  int idx = blockIdx.x * 256 + threadIdx.x;           // over B*L
  if (idx < B_ * L_) {
    int b = idx / L_, l = idx % L_;
    out_zmask[idx] = mask[(size_t)b * T_ + l * STRIDE_] ? 1.f : 0.f;
  }
  if (idx < B_) out_zlen[idx] = ceilf((float)xlen[idx] / (float)STRIDE_);
  if (idx == 0) {
    float s = 0.f;
    for (int b = 0; b < B_; ++b) s += loss_b[b];
    out_loss[0] = s / (float)B_;
  }
}

// ---------------------------------------------------------------------------
// 5) z[b,d,l] = sum_t align[b,l,t] * x[b,d,t]   (WMMA f16->f32)
//    block = 256 threads = 8 waves; tile 32(d) x 64(l); K chunks of 32.
//    A frag: x rows (16x32 f16), B frag: align rows used as KxN (32x16 f16).
// ---------------------------------------------------------------------------
__global__ __launch_bounds__(256) void k_gemm(const float* __restrict__ x,
                                              const float* __restrict__ align,
                                              float* __restrict__ z) {
  __shared__ _Float16 lx[32][34];  // x tile:     32 d-rows x 32 t
  __shared__ _Float16 la[64][34];  // align tile: 64 l-rows x 32 t

  int bid = blockIdx.x;            // B * 16 * 16 = 2048 blocks
  int b   = bid >> 8;
  int rem = bid & 255;
  int dt  = rem >> 4, lt = rem & 15;
  int d0  = dt * 32,  l0 = lt * 64;

  int tid  = threadIdx.x;
  int lane = tid & 31, w = tid >> 5;
  int wm   = w >> 2,  wn = w & 3;          // 2 x 4 waves of 16x16 tiles
  int half = lane >> 4, mr = lane & 15;

  const float* xp = x     + ((size_t)b * D_ + d0) * T_;
  const float* ap = align + ((size_t)b * L_ + l0) * T_;

  int srow = tid >> 3;             // 0..31
  int scol = (tid & 7) * 4;        // 0..28

  v8f c = {0.f, 0.f, 0.f, 0.f, 0.f, 0.f, 0.f, 0.f};

  for (int k0 = 0; k0 < T_; k0 += 32) {
    f4 vx  = *(const f4*)(xp + (size_t)srow * T_ + k0 + scol);
    f4 va0 = *(const f4*)(ap + (size_t)srow * T_ + k0 + scol);
    f4 va1 = *(const f4*)(ap + (size_t)(srow + 32) * T_ + k0 + scol);
    __syncthreads();               // previous iteration's readers done
    lx[srow][scol + 0] = (_Float16)vx.x;
    lx[srow][scol + 1] = (_Float16)vx.y;
    lx[srow][scol + 2] = (_Float16)vx.z;
    lx[srow][scol + 3] = (_Float16)vx.w;
    la[srow][scol + 0] = (_Float16)va0.x;
    la[srow][scol + 1] = (_Float16)va0.y;
    la[srow][scol + 2] = (_Float16)va0.z;
    la[srow][scol + 3] = (_Float16)va0.w;
    la[srow + 32][scol + 0] = (_Float16)va1.x;
    la[srow + 32][scol + 1] = (_Float16)va1.y;
    la[srow + 32][scol + 2] = (_Float16)va1.z;
    la[srow + 32][scol + 3] = (_Float16)va1.w;
    __syncthreads();

    // Gather fragments per documented 16-bit WMMA layouts (wave32).
    v16h af, bf;
#pragma unroll
    for (int v = 0; v < 8; ++v) {
      // A 16x32: VGPR v<4 -> K = half*8 + 2v ; v>=4 -> K = 16 + half*8 + 2(v-4)
      int ka = (v < 4) ? (half * 8 + 2 * v) : (16 + half * 8 + 2 * (v - 4));
      h2 pa = *(const h2*)&lx[wm * 16 + mr][ka];
      af[2 * v]     = pa.x;
      af[2 * v + 1] = pa.y;
      // B 32x16: lane n=mr holds column n; VGPR v<4 -> K = half*8 + 2v ; v>=4 -> K = 16 + half*8 + 2(v-4)
      int kb = (v < 4) ? (half * 8 + 2 * v) : (16 + half * 8 + 2 * (v - 4));
      h2 pb = *(const h2*)&la[wn * 16 + mr][kb];
      bf[2 * v]     = pb.x;
      bf[2 * v + 1] = pb.y;
    }
    c = __builtin_amdgcn_wmma_f32_16x16x32_f16(false, af, false, bf,
                                               (short)0, c, false, false);
  }

  // C/D 16x16 f32: VGPR r -> M = r + 8*half; N = mr
  float* zp = z + ((size_t)b * D_ + d0 + wm * 16) * L_ + l0 + wn * 16 + mr;
#pragma unroll
  for (int r = 0; r < 8; ++r) {
    int m = r + 8 * half;
    zp[(size_t)m * L_] = c[r];
  }
}

// ---------------------------------------------------------------------------
extern "C" void kernel_launch(void* const* d_in, const int* in_sizes, int n_in,
                              void* d_out, int out_size, void* d_ws, size_t ws_size,
                              hipStream_t stream) {
  const float*   x    = (const float*)d_in[0];     // (B,D,T)
  const float*   W    = (const float*)d_in[1];     // (1,D)
  const uint8_t* mask = (const uint8_t*)d_in[2];   // (B,T) bool
  const int*     xlen = (const int*)d_in[3];       // (B,)
  float* out = (float*)d_out;

  float* ws_score = (float*)d_ws;                  // B*T
  float* ws_csn   = ws_score + (size_t)B_ * T_;    // B*T
  float* ws_loss  = ws_csn + (size_t)B_ * T_;      // B

  float* out_z     = out + OFF_Z;
  float* out_zmask = out + OFF_ZMASK;
  float* out_zlen  = out + OFF_ZLEN;
  float* out_align = out + OFF_ALIGN;
  float* out_loss  = out + OFF_LOSS;

  k_score<<<(B_ * T_) / 256, 256, 0, stream>>>(x, W, mask, ws_score);
  k_scan<<<B_, 1024, 0, stream>>>(ws_score, xlen, ws_csn, ws_loss);
  k_align<<<B_ * L_, 256, 0, stream>>>(ws_csn, mask, out_align);
  k_misc<<<(B_ * L_) / 256, 256, 0, stream>>>(mask, xlen, ws_loss,
                                              out_zmask, out_zlen, out_loss);
  k_gemm<<<B_ * 16 * 16, 256, 0, stream>>>(x, out_align, out_z);
}
